// BatchInvariantAttention_81381040324777
// MI455X (gfx1250) — compile-verified, matrix operand order checked by reading, not run on hardware
//
#include <hip/hip_runtime.h>

// ---------------------------------------------------------------------------
// MHA forward for MI455X (gfx1250): bf16 WMMA + TDM (tensor_load_to_lds)
// double-buffered flash attention, ping-pong software-pipelined WMMA GEMMs.
// ---------------------------------------------------------------------------

typedef __bf16 bf16;
typedef __attribute__((ext_vector_type(8)))  bf16  v8bf;
typedef __attribute__((ext_vector_type(16))) bf16  v16bf;
typedef __attribute__((ext_vector_type(8)))  float v8f;
typedef __attribute__((ext_vector_type(4)))  unsigned int u32x4;
typedef __attribute__((ext_vector_type(8)))  int i32x8;
typedef __attribute__((ext_vector_type(4)))  int i32x4;

#define HIDDEN   1024
#define HEADS    16
#define HEAD_DIM 64
#define BATCH    2
#define SEQ      2048
#define TOKENS   (BATCH * SEQ)

// Per-lane A/B fragment: two contiguous 16-byte runs (CDNA5 16-bit layout:
// dwords 0-3 <-> K = half*8+0..7, dwords 4-7 <-> K = 16+half*8+0..7).
__device__ __forceinline__ v16bf ld_frag(const bf16* p_lo, const bf16* p_hi) {
    v8bf lo = *reinterpret_cast<const v8bf*>(p_lo);
    v8bf hi = *reinterpret_cast<const v8bf*>(p_hi);
    return __builtin_shufflevector(lo, hi, 0, 1, 2, 3, 4, 5, 6, 7,
                                           8, 9, 10, 11, 12, 13, 14, 15);
}

__device__ __forceinline__ v8f wmma_bf16(v16bf a, v16bf b, v8f c) {
    return __builtin_amdgcn_wmma_f32_16x16x32_bf16(false, a, false, b,
                                                   (short)0, c, false, false);
}

// ---------------------------------------------------------------------------
// TDM: async 2D bf16 tile load Global -> LDS (descriptor per
// CDNA5 ISA). 6-arg builtin on this toolchain.
// ---------------------------------------------------------------------------
__device__ __forceinline__ void tdm_load_2d_bf16(unsigned lds_off,
                                                 const bf16* gptr,
                                                 unsigned tile_w,   // elems
                                                 unsigned tile_h,   // rows
                                                 unsigned row_stride /* elems */) {
    unsigned long long ga = (unsigned long long)(const void*)gptr;
    u32x4 g0;
    g0.x = 1u;                                         // count=1, user D#
    g0.y = lds_off;                                    // lds_addr (bytes)
    g0.z = (unsigned)ga;                               // global_addr[31:0]
    g0.w = (unsigned)((ga >> 32) & 0x01ffffffu)        // global_addr[56:32]
           | (2u << 30);                               // type = 2 ("image")
    i32x8 g1;
    g1[0] = (int)(1u << 16);                           // data_size = 1 (2B)
    g1[1] = (int)((tile_w & 0xffffu) << 16);           // tensor_dim0[15:0]
    g1[2] = (int)((tile_w >> 16) | ((tile_h & 0xffffu) << 16)); // d0 hi | d1 lo
    g1[3] = (int)((tile_h >> 16) | ((tile_w & 0xffffu) << 16)); // d1 hi | tile_dim0
    g1[4] = (int)(tile_h & 0xffffu);                   // tile_dim1 (tile_dim2=0)
    g1[5] = (int)row_stride;                           // tensor_dim0_stride[31:0]
    g1[6] = 0;
    g1[7] = 0;
    i32x4 z4 = {0, 0, 0, 0};
    i32x8 z8 = {0, 0, 0, 0, 0, 0, 0, 0};
    __builtin_amdgcn_tensor_load_to_lds(g0, g1, z4, z4, z8, 0);
}

// Low 32 bits of a flat LDS address are the LDS byte offset (ISA §10.2).
__device__ __forceinline__ unsigned lds_offset(const void* p) {
    return (unsigned)(size_t)p;
}

// ---------------------------------------------------------------------------
// fp32 -> bf16 conversion
// ---------------------------------------------------------------------------
__global__ void k_f32_to_bf16(const float* __restrict__ src,
                              bf16* __restrict__ dst, int n) {
    int i = (blockIdx.x * blockDim.x + threadIdx.x) * 4;
    if (i + 3 < n) {
        float4 f = *reinterpret_cast<const float4*>(src + i);
        dst[i + 0] = (bf16)f.x;
        dst[i + 1] = (bf16)f.y;
        dst[i + 2] = (bf16)f.z;
        dst[i + 3] = (bf16)f.w;
    }
}

// ---------------------------------------------------------------------------
// GEMM core: 32(M) x 64(N) per wave, Y = X @ W^T. Ping-pong pipelined over K:
// two named fragment sets alternate roles so no rotation copies are emitted.
// Subtile strides are compile-time immediates; only 2 pointers stay live.
// ---------------------------------------------------------------------------
#define RSTRIDE (16 * HIDDEN)   // 16 rows, in elements (32 KiB in bytes)

struct GemmAcc { v8f a[2][4]; };

__device__ __forceinline__ void gemm_load(const bf16* ap, const bf16* bp,
                                          v16bf& A0, v16bf& A1,
                                          v16bf& B0, v16bf& B1,
                                          v16bf& B2, v16bf& B3) {
    A0 = ld_frag(ap,           ap + 16);
    A1 = ld_frag(ap + RSTRIDE, ap + RSTRIDE + 16);
    B0 = ld_frag(bp,               bp + 16);
    B1 = ld_frag(bp + RSTRIDE,     bp + RSTRIDE + 16);
    B2 = ld_frag(bp + 2 * RSTRIDE, bp + 2 * RSTRIDE + 16);
    B3 = ld_frag(bp + 3 * RSTRIDE, bp + 3 * RSTRIDE + 16);
}

__device__ __forceinline__ void gemm_mmas(GemmAcc& g,
                                          v16bf A0, v16bf A1,
                                          v16bf B0, v16bf B1,
                                          v16bf B2, v16bf B3) {
    g.a[0][0] = wmma_bf16(A0, B0, g.a[0][0]);
    g.a[0][1] = wmma_bf16(A0, B1, g.a[0][1]);
    g.a[0][2] = wmma_bf16(A0, B2, g.a[0][2]);
    g.a[0][3] = wmma_bf16(A0, B3, g.a[0][3]);
    g.a[1][0] = wmma_bf16(A1, B0, g.a[1][0]);
    g.a[1][1] = wmma_bf16(A1, B1, g.a[1][1]);
    g.a[1][2] = wmma_bf16(A1, B2, g.a[1][2]);
    g.a[1][3] = wmma_bf16(A1, B3, g.a[1][3]);
}

__device__ __forceinline__ GemmAcc gemm_32x64(const bf16* __restrict__ X,
                                              const bf16* __restrict__ W,
                                              int m0, int nbase,
                                              int lm, int half) {
    const bf16* ap = X + (size_t)(m0 + lm) * HIDDEN + half * 8;
    const bf16* bp = W + (size_t)(nbase + lm) * HIDDEN + half * 8;

    GemmAcc g = {};
    v16bf Aa0, Aa1, Ba0, Ba1, Ba2, Ba3;   // set a
    v16bf Ab0, Ab1, Bb0, Bb1, Bb2, Bb3;   // set b

    gemm_load(ap, bp, Aa0, Aa1, Ba0, Ba1, Ba2, Ba3);          // k = 0

    #pragma unroll 1
    for (int k = 0; k < HIDDEN - 64; k += 64) {
        gemm_load(ap + 32, bp + 32, Ab0, Ab1, Bb0, Bb1, Bb2, Bb3);  // k+32
        gemm_mmas(g, Aa0, Aa1, Ba0, Ba1, Ba2, Ba3);                 // k
        gemm_load(ap + 64, bp + 64, Aa0, Aa1, Ba0, Ba1, Ba2, Ba3);  // k+64
        gemm_mmas(g, Ab0, Ab1, Bb0, Bb1, Bb2, Bb3);                 // k+32
        ap += 64; bp += 64;
    }
    // tail: set a holds k = HIDDEN-64
    gemm_load(ap + 32, bp + 32, Ab0, Ab1, Bb0, Bb1, Bb2, Bb3);      // HIDDEN-32
    gemm_mmas(g, Aa0, Aa1, Ba0, Ba1, Ba2, Ba3);
    gemm_mmas(g, Ab0, Ab1, Bb0, Bb1, Bb2, Bb3);
    return g;
}

// writeVt != 0: scatter the V projection transposed per head: Vt[b][h][hd][s]
__global__ __launch_bounds__(32, 1)
void k_proj_gemm(const bf16* __restrict__ X, const bf16* __restrict__ W,
                 bf16* __restrict__ Y, int writeVt) {
    const int lane = threadIdx.x;
    const int lm   = lane & 15;
    const int half = lane >> 4;
    const int m0    = blockIdx.x * 32;
    const int nbase = blockIdx.y * 64;

    GemmAcc g = gemm_32x64(X, W, m0, nbase, lm, half);

    #pragma unroll
    for (int m = 0; m < 2; ++m) {
        #pragma unroll
        for (int t = 0; t < 4; ++t) {
            #pragma unroll
            for (int r = 0; r < 8; ++r) {
                int rowg = m0 + m * 16 + half * 8 + r;
                int colg = nbase + t * 16 + lm;
                bf16 v = (bf16)g.a[m][t][r];
                if (!writeVt) {
                    Y[(size_t)rowg * HIDDEN + colg] = v;
                } else {
                    int bb = rowg >> 11, s = rowg & (SEQ - 1);
                    int hh = colg >> 6,  hd = colg & (HEAD_DIM - 1);
                    Y[(size_t)(((bb * HEADS + hh) * HEAD_DIM) + hd) * SEQ + s] = v;
                }
            }
        }
    }
}

__global__ __launch_bounds__(32, 1)
void k_out_gemm(const bf16* __restrict__ X, const bf16* __restrict__ W,
                float* __restrict__ Y) {
    const int lane = threadIdx.x;
    const int lm   = lane & 15;
    const int half = lane >> 4;
    const int m0    = blockIdx.x * 32;
    const int nbase = blockIdx.y * 64;

    GemmAcc g = gemm_32x64(X, W, m0, nbase, lm, half);

    #pragma unroll
    for (int m = 0; m < 2; ++m)
        #pragma unroll
        for (int t = 0; t < 4; ++t)
            #pragma unroll
            for (int r = 0; r < 8; ++r) {
                int rowg = m0 + m * 16 + half * 8 + r;
                int colg = nbase + t * 16 + lm;
                Y[(size_t)rowg * HIDDEN + colg] = g.a[m][t][r];
            }
}

// ---------------------------------------------------------------------------
// Flash attention, TDM double-buffered. One wave per (b, h, 16-query tile).
// ---------------------------------------------------------------------------
__global__ __launch_bounds__(32, 1)
void k_attention(const bf16* __restrict__ Qb,
                 const bf16* __restrict__ Kb,
                 const bf16* __restrict__ Vt,
                 const float* __restrict__ mask,
                 bf16* __restrict__ Ob) {
    const int lane = threadIdx.x;
    const int lm   = lane & 15;
    const int half = lane >> 4;
    const int qt = blockIdx.x, h = blockIdx.y, b = blockIdx.z;

    __shared__ bf16 Klds[2][32 * 64];   // [key-local][hd]      4 KB x2
    __shared__ bf16 Vlds[2][64 * 32];   // [hd][s-local]        4 KB x2
    __shared__ bf16 Plds[16 * 32];      // P tile C/D -> A layout handoff

    const bf16* qp = Qb + (size_t)(b * SEQ + qt * 16 + lm) * HIDDEN
                        + h * HEAD_DIM + half * 8;
    v16bf q0 = ld_frag(qp,      qp + 16);
    v16bf q1 = ld_frag(qp + 32, qp + 48);

    float mrun[8], lrun[8];
    v8f o0 = {}, o1 = {}, o2 = {}, o3 = {};
    #pragma unroll
    for (int r = 0; r < 8; ++r) { mrun[r] = -1e30f; lrun[r] = 0.0f; }

    const float* mbase = mask + (size_t)b * SEQ * SEQ;
    const bf16*  khead = Kb + (size_t)(b * SEQ) * HIDDEN + h * HEAD_DIM;
    const bf16*  vhead = Vt + (size_t)((b * HEADS + h) * HEAD_DIM) * SEQ;

    // prologue: chunk 0 into buffer 0
    tdm_load_2d_bf16(lds_offset(&Klds[0][0]), khead, HEAD_DIM, 32, HIDDEN);
    tdm_load_2d_bf16(lds_offset(&Vlds[0][0]), vhead, 32, HEAD_DIM, SEQ);

    const int NCHUNK = SEQ / 32;
    for (int ci = 0; ci < NCHUNK; ++ci) {
        const int kb = ci * 32;
        if (ci + 1 < NCHUNK) {
            const int buf = (ci + 1) & 1;
            tdm_load_2d_bf16(lds_offset(&Klds[buf][0]),
                             khead + (size_t)(kb + 32) * HIDDEN, HEAD_DIM, 32, HIDDEN);
            tdm_load_2d_bf16(lds_offset(&Vlds[buf][0]),
                             vhead + (kb + 32), 32, HEAD_DIM, SEQ);
            __builtin_amdgcn_s_wait_tensorcnt((short)2);  // current buffer done
        } else {
            __builtin_amdgcn_s_wait_tensorcnt((short)0);
        }
        const bf16* kp = &Klds[ci & 1][0] + lm * HEAD_DIM + half * 8;
        const bf16* vp = &Vlds[ci & 1][0] + lm * 32 + half * 8;

        // ---- scores: two 16-key subtiles, B frags from LDS -------------
        v8f s0 = {}, s1 = {};
        s0 = wmma_bf16(q0, ld_frag(kp,      kp + 16), s0);
        s0 = wmma_bf16(q1, ld_frag(kp + 32, kp + 48), s0);
        s1 = wmma_bf16(q0, ld_frag(kp + 16 * HEAD_DIM,      kp + 16 * HEAD_DIM + 16), s1);
        s1 = wmma_bf16(q1, ld_frag(kp + 16 * HEAD_DIM + 32, kp + 16 * HEAD_DIM + 48), s1);

        // ---- scale + mask + online softmax -----------------------------
        #pragma unroll
        for (int r = 0; r < 8; ++r) {
            const int rowq = qt * 16 + half * 8 + r;
            float x0 = s0[r] * 0.125f + mbase[(size_t)rowq * SEQ + kb + lm];
            float x1 = s1[r] * 0.125f + mbase[(size_t)rowq * SEQ + kb + 16 + lm];

            float cm = fmaxf(x0, x1);
            #pragma unroll
            for (int off = 1; off < 16; off <<= 1)
                cm = fmaxf(cm, __shfl_xor(cm, off, 32));

            float nm    = fmaxf(mrun[r], cm);
            float alpha = __expf(mrun[r] - nm);
            mrun[r] = nm;

            float p0 = __expf(x0 - nm);
            float p1 = __expf(x1 - nm);
            float rs = p0 + p1;
            #pragma unroll
            for (int off = 1; off < 16; off <<= 1)
                rs += __shfl_xor(rs, off, 32);
            lrun[r] = lrun[r] * alpha + rs;

            o0[r] *= alpha; o1[r] *= alpha; o2[r] *= alpha; o3[r] *= alpha;

            Plds[(half * 8 + r) * 32 + lm]      = (bf16)p0;
            Plds[(half * 8 + r) * 32 + 16 + lm] = (bf16)p1;
        }
        __syncthreads();   // 1-wave group: ordering pin (DS in-order per wave)

        // ---- P (16x32, A layout) x V chunk (B frags from LDS) ----------
        const bf16* pp = &Plds[lm * 32 + half * 8];
        v16bf pa = ld_frag(pp, pp + 16);
        o0 = wmma_bf16(pa, ld_frag(vp,        vp + 16),        o0);
        o1 = wmma_bf16(pa, ld_frag(vp + 512,  vp + 512 + 16),  o1);
        o2 = wmma_bf16(pa, ld_frag(vp + 1024, vp + 1024 + 16), o2);
        o3 = wmma_bf16(pa, ld_frag(vp + 1536, vp + 1536 + 16), o3);
        __syncthreads();
    }

    // ---- normalize + store [b, s, h*hd] for final GEMM ------------------
    #pragma unroll
    for (int r = 0; r < 8; ++r) {
        float inv = 1.0f / lrun[r];
        int token = b * SEQ + qt * 16 + half * 8 + r;
        bf16* orow = Ob + (size_t)token * HIDDEN + h * HEAD_DIM;
        orow[ 0 + lm] = (bf16)(o0[r] * inv);
        orow[16 + lm] = (bf16)(o1[r] * inv);
        orow[32 + lm] = (bf16)(o2[r] * inv);
        orow[48 + lm] = (bf16)(o3[r] * inv);
    }
}

// ---------------------------------------------------------------------------
// Host launch
// ---------------------------------------------------------------------------
extern "C" void kernel_launch(void* const* d_in, const int* in_sizes, int n_in,
                              void* d_out, int out_size, void* d_ws, size_t ws_size,
                              hipStream_t stream) {
    (void)in_sizes; (void)n_in; (void)out_size; (void)ws_size;

    const float* hs   = (const float*)d_in[0];
    const float* mask = (const float*)d_in[1];
    const float* wq   = (const float*)d_in[2];
    const float* wk   = (const float*)d_in[3];
    const float* wv   = (const float*)d_in[4];
    const float* wo   = (const float*)d_in[5];
    float* out = (float*)d_out;

    char* ws = (char*)d_ws;
    const size_t MB = 1u << 20;
    bf16* Xb  = (bf16*)(ws +  0 * MB);
    bf16* Wqb = (bf16*)(ws +  8 * MB);
    bf16* Wkb = (bf16*)(ws + 10 * MB);
    bf16* Wvb = (bf16*)(ws + 12 * MB);
    bf16* Wob = (bf16*)(ws + 14 * MB);
    bf16* Qb  = (bf16*)(ws + 16 * MB);
    bf16* Kb  = (bf16*)(ws + 24 * MB);
    bf16* Vt  = (bf16*)(ws + 32 * MB);
    bf16* Ob  = (bf16*)(ws + 40 * MB);

    const int nX = TOKENS * HIDDEN;
    const int nW = HIDDEN * HIDDEN;
    dim3 cb(256);
    k_f32_to_bf16<<<dim3((nX / 4 + 255) / 256), cb, 0, stream>>>(hs, Xb, nX);
    k_f32_to_bf16<<<dim3((nW / 4 + 255) / 256), cb, 0, stream>>>(wq, Wqb, nW);
    k_f32_to_bf16<<<dim3((nW / 4 + 255) / 256), cb, 0, stream>>>(wk, Wkb, nW);
    k_f32_to_bf16<<<dim3((nW / 4 + 255) / 256), cb, 0, stream>>>(wv, Wvb, nW);
    k_f32_to_bf16<<<dim3((nW / 4 + 255) / 256), cb, 0, stream>>>(wo, Wob, nW);

    dim3 gproj(TOKENS / 32, HIDDEN / 64);   // 128 x 16 waves
    dim3 blk(32);
    k_proj_gemm<<<gproj, blk, 0, stream>>>(Xb, Wqb, Qb, 0);
    k_proj_gemm<<<gproj, blk, 0, stream>>>(Xb, Wkb, Kb, 0);
    k_proj_gemm<<<gproj, blk, 0, stream>>>(Xb, Wvb, Vt, 1);   // transposed V

    dim3 gattn(SEQ / 16, HEADS, BATCH);     // 128 x 16 x 2 waves
    k_attention<<<gattn, blk, 0, stream>>>(Qb, Kb, Vt, mask, Ob);

    k_out_gemm<<<gproj, blk, 0, stream>>>(Ob, Wob, out);
}
